// Mini_pointgnn_v9_67310727463243
// MI455X (gfx1250) — matrix operand order repaired and checked by reading, not compiled, not run on hardware
//
#include <hip/hip_runtime.h>

// ---------------------------------------------------------------------------
// Mini PointGNN pipeline for gfx1250 (MI455X), wave32 + WMMA f16->f32.
// Memory-bound workload: fuse gather -> MLP(WMMA) -> scatter per stage.
// ---------------------------------------------------------------------------

#define WPB 4  // waves per block (128 threads)

typedef __attribute__((ext_vector_type(16))) _Float16 v16h;
typedef __attribute__((ext_vector_type(8)))  float    v8f;

__device__ __forceinline__ v8f zero8() {
  v8f z = {0.f,0.f,0.f,0.f,0.f,0.f,0.f,0.f};
  return z;
}

__device__ __forceinline__ v8f wmma16(v16h a, v16h b, v8f c) {
  return __builtin_amdgcn_wmma_f32_16x16x32_f16(false, a, false, b,
                                                (short)0, c, false, false);
}

// A-fragment (16x32 f16 tile) per CDNA5 ISA layout:
// lanes 0-15: rows 0-15, VGPR j holds K pairs {0,1},{2,3},{4,5},{6,7},{16..23}
// lanes 16-31: same rows, K pattern +8.
__device__ __forceinline__ v16h load_afrag(const _Float16* x, int stride, int c) {
  int l = threadIdx.x & 31;
  int r = l & 15, hf = l >> 4;
  union { v16h v; unsigned u[8]; } t;
  const unsigned* row = (const unsigned*)(x + r * stride);
  int kb = c * 32 + 8 * hf;
#pragma unroll
  for (int j = 0; j < 8; ++j) {
    int k = kb + ((j < 4) ? (2 * j) : (16 + 2 * (j - 4)));
    t.u[j] = row[k >> 1];
  }
  return t.v;
}

__device__ __forceinline__ void store4h(_Float16* dst, float a, float b,
                                        float c, float d) {
  union { unsigned long long u; _Float16 h[4]; } t;
  t.h[0] = (_Float16)a; t.h[1] = (_Float16)b;
  t.h[2] = (_Float16)c; t.h[3] = (_Float16)d;
  *(unsigned long long*)dst = t.u;  // 8B-aligned ds_store_b64
}

// Two-layer MLP on a 16-row tile, one wave. x16: LDS f16 (16 x xstr),
// h16: LDS f16 scratch (16 x 64). Weights pre-packed into B-frag layout.
template <int CH1>
__device__ __forceinline__ void mlp2(const _Float16* x16, int xstr,
                                     _Float16* h16,
                                     const _Float16* wf1, const float* b1,
                                     const _Float16* wf2, const float* b2,
                                     v8f y[4]) {
  int l = threadIdx.x & 31;
  int n = l & 15, hf = l >> 4;
  v8f acc[4];
#pragma unroll
  for (int t = 0; t < 4; ++t) acc[t] = zero8();
#pragma unroll
  for (int c = 0; c < CH1; ++c) {
    v16h a = load_afrag(x16, xstr, c);
#pragma unroll
    for (int t = 0; t < 4; ++t) {
      v16h b = *(const v16h*)(wf1 + (((c * 4 + t) * 32 + l) << 4));
      acc[t] = wmma16(a, b, acc[t]);
    }
  }
  // bias + relu, stage H into LDS as next layer's A matrix
#pragma unroll
  for (int t = 0; t < 4; ++t) {
    float bb = b1[t * 16 + n];
#pragma unroll
    for (int i = 0; i < 8; ++i)
      h16[(i + 8 * hf) * 64 + t * 16 + n] = (_Float16)fmaxf(acc[t][i] + bb, 0.f);
  }
  asm volatile("s_wait_dscnt 0x0" ::: "memory");
#pragma unroll
  for (int t = 0; t < 4; ++t) y[t] = zero8();
#pragma unroll
  for (int c = 0; c < 2; ++c) {
    v16h a = load_afrag(h16, 64, c);
#pragma unroll
    for (int t = 0; t < 4; ++t) {
      v16h b = *(const v16h*)(wf2 + (((c * 4 + t) * 32 + l) << 4));
      y[t] = wmma16(a, b, y[t]);
    }
  }
#pragma unroll
  for (int t = 0; t < 4; ++t) {
    float bb = b2[t * 16 + n];
#pragma unroll
    for (int i = 0; i < 8; ++i) y[t][i] = fmaxf(y[t][i] + bb, 0.f);
  }
}

// ---- weight packing into CDNA5 B-fragment layout: k = c*32+(e<8 ? 8*half+e : 16+8*half+e-8) -----
__global__ void k_pack(const float* __restrict__ W, _Float16* __restrict__ dst,
                       int K, int N, int chunks, int ntiles) {
  int id = blockIdx.x * blockDim.x + threadIdx.x;
  int total = chunks * ntiles * 512;
  if (id >= total) return;
  int e = id & 15;
  int l = (id >> 4) & 31;
  int t = (id >> 9) % ntiles;
  int c = id / (512 * ntiles);
  int n = t * 16 + (l & 15);
  int hf = l >> 4;
  int k = c * 32 + ((e < 8) ? (hf * 8 + e) : (16 + hf * 8 + (e - 8)));
  _Float16 v = (_Float16)0.f;
  if (k < K && n < N) v = (_Float16)W[k * N + n];
  dst[id] = v;
}

// ---- stage 1: ffn MLP per point + atomic segment_sum into f1 -------------
__global__ void k_ffn(const float* __restrict__ pts, const float* __restrict__ rem,
                      const float* __restrict__ l1c, const int* __restrict__ l1l,
                      int N, const _Float16* wf1, const float* b1,
                      const _Float16* wf2, const float* b2, float* f1) {
  __shared__ _Float16 xs[WPB][16 * 32];
  __shared__ _Float16 hs[WPB][16 * 64];
  __shared__ int lab[WPB][16];
  int w = threadIdx.x >> 5;
  int tile = blockIdx.x * WPB + w;
  if (tile * 16 >= N) return;
  int l = threadIdx.x & 31, r = l & 15, hf = l >> 4;
  int idx = tile * 16 + r;
  bool valid = idx < N;
  int ic = valid ? idx : N - 1;
  if (hf == 0) {
    int li = l1l[ic];
    lab[w][r] = valid ? li : -1;
    float x0 = pts[ic * 3 + 0] - l1c[li * 3 + 0];
    float x1 = pts[ic * 3 + 1] - l1c[li * 3 + 1];
    float x2 = pts[ic * 3 + 2] - l1c[li * 3 + 2];
    float x3 = rem[ic];
    _Float16* xr = &xs[w][r * 32];
    store4h(xr, x0, x1, x2, x3);
    store4h(xr + 4, 0, 0, 0, 0);
    store4h(xr + 8, 0, 0, 0, 0);
    store4h(xr + 12, 0, 0, 0, 0);
  } else {
    _Float16* xr = &xs[w][r * 32 + 16];
    store4h(xr, 0, 0, 0, 0);      store4h(xr + 4, 0, 0, 0, 0);
    store4h(xr + 8, 0, 0, 0, 0);  store4h(xr + 12, 0, 0, 0, 0);
  }
  asm volatile("s_wait_dscnt 0x0" ::: "memory");
  v8f y[4];
  mlp2<1>(xs[w], 32, hs[w], wf1, b1, wf2, b2, y);
  int n = l & 15;
#pragma unroll
  for (int i = 0; i < 8; ++i) {
    int d = lab[w][i + 8 * hf];
    if (d >= 0) {
#pragma unroll
      for (int t = 0; t < 4; ++t)
        atomicAdd(&f1[d * 64 + t * 16 + n], y[t][i]);
    }
  }
}

// ---- edge MLP + atomic segment_max (values >= 0 post-relu) ---------------
__global__ void k_edge(const float* __restrict__ feat, const float* __restrict__ cen,
                       const int* __restrict__ edges, int M,
                       const _Float16* wf1, const float* b1,
                       const _Float16* wf2, const float* b2, float* agg) {
  __shared__ _Float16 xs[WPB][16 * 96];
  __shared__ _Float16 hs[WPB][16 * 64];
  __shared__ int lab[WPB][16];
  int w = threadIdx.x >> 5;
  int tile = blockIdx.x * WPB + w;
  if (tile * 16 >= M) return;
  int l = threadIdx.x & 31, r = l & 15, hf = l >> 4;
  int e = tile * 16 + r;
  bool valid = e < M;
  int ec = valid ? e : M - 1;
  int s = edges[2 * ec], d = edges[2 * ec + 1];
  const float4* fp = (const float4*)(feat + s * 64) + hf * 8;
  _Float16* xr = &xs[w][r * 96 + hf * 32];
#pragma unroll
  for (int q = 0; q < 8; ++q) {
    float4 v = fp[q];
    store4h(xr + 4 * q, v.x, v.y, v.z, v.w);
  }
  if (hf == 0) {
    lab[w][r] = valid ? d : -1;
    float rx = cen[s * 3 + 0] - cen[d * 3 + 0];
    float ry = cen[s * 3 + 1] - cen[d * 3 + 1];
    float rz = cen[s * 3 + 2] - cen[d * 3 + 2];
    _Float16* xt = &xs[w][r * 96 + 64];
    store4h(xt, rx, ry, rz, 0.f);
    store4h(xt + 4, 0, 0, 0, 0);
    store4h(xt + 8, 0, 0, 0, 0);
    store4h(xt + 12, 0, 0, 0, 0);
  } else {
    _Float16* xt = &xs[w][r * 96 + 80];
    store4h(xt, 0, 0, 0, 0);      store4h(xt + 4, 0, 0, 0, 0);
    store4h(xt + 8, 0, 0, 0, 0);  store4h(xt + 12, 0, 0, 0, 0);
  }
  asm volatile("s_wait_dscnt 0x0" ::: "memory");
  v8f y[4];
  mlp2<3>(xs[w], 96, hs[w], wf1, b1, wf2, b2, y);
  int n = l & 15;
#pragma unroll
  for (int i = 0; i < 8; ++i) {
    int dd = lab[w][i + 8 * hf];
    if (dd >= 0) {
#pragma unroll
      for (int t = 0; t < 4; ++t)
        atomicMax((int*)&agg[dd * 64 + t * 16 + n], __float_as_int(y[t][i]));
    }
  }
}

// ---- GNN output MLP: out = feat + mlp(agg) (+res) ------------------------
__global__ void k_gnnout(const float* __restrict__ feat, const float* __restrict__ agg,
                         int N, const _Float16* wf1, const float* b1,
                         const _Float16* wf2, const float* b2,
                         const float* __restrict__ res, float* out) {
  __shared__ _Float16 xs[WPB][16 * 64];
  __shared__ _Float16 hs[WPB][16 * 64];
  int w = threadIdx.x >> 5;
  int tile = blockIdx.x * WPB + w;
  if (tile * 16 >= N) return;
  int l = threadIdx.x & 31, r = l & 15, hf = l >> 4;
  int idx = tile * 16 + r;
  int ic = idx < N ? idx : N - 1;
  const float4* fp = (const float4*)(agg + ic * 64) + hf * 8;
  _Float16* xr = &xs[w][r * 64 + hf * 32];
#pragma unroll
  for (int q = 0; q < 8; ++q) {
    float4 v = fp[q];
    store4h(xr + 4 * q, v.x, v.y, v.z, v.w);
  }
  asm volatile("s_wait_dscnt 0x0" ::: "memory");
  v8f y[4];
  mlp2<2>(xs[w], 64, hs[w], wf1, b1, wf2, b2, y);
  int n = l & 15;
#pragma unroll
  for (int i = 0; i < 8; ++i) {
    int row = tile * 16 + i + 8 * hf;
    if (row < N) {
#pragma unroll
      for (int t = 0; t < 4; ++t) {
        int col = t * 16 + n;
        float v = feat[row * 64 + col] + y[t][i];
        if (res) v += res[row * 64 + col];
        out[row * 64 + col] = v;
      }
    }
  }
}

// ---- m2l: per-l1-node MLP([f2, l1c - l2c[lab]]) + seg_max into f3 --------
__global__ void k_m2l(const float* __restrict__ feat, const float* __restrict__ l1c,
                      const float* __restrict__ l2c, const int* __restrict__ l2l,
                      int N, const _Float16* wf1, const float* b1,
                      const _Float16* wf2, const float* b2, float* f3) {
  __shared__ _Float16 xs[WPB][16 * 96];
  __shared__ _Float16 hs[WPB][16 * 64];
  __shared__ int lab[WPB][16];
  int w = threadIdx.x >> 5;
  int tile = blockIdx.x * WPB + w;
  if (tile * 16 >= N) return;
  int l = threadIdx.x & 31, r = l & 15, hf = l >> 4;
  int idx = tile * 16 + r;
  bool valid = idx < N;
  int ic = valid ? idx : N - 1;
  int g = l2l[ic];
  const float4* fp = (const float4*)(feat + ic * 64) + hf * 8;
  _Float16* xr = &xs[w][r * 96 + hf * 32];
#pragma unroll
  for (int q = 0; q < 8; ++q) {
    float4 v = fp[q];
    store4h(xr + 4 * q, v.x, v.y, v.z, v.w);
  }
  if (hf == 0) {
    lab[w][r] = valid ? g : -1;
    float rx = l1c[ic * 3 + 0] - l2c[g * 3 + 0];
    float ry = l1c[ic * 3 + 1] - l2c[g * 3 + 1];
    float rz = l1c[ic * 3 + 2] - l2c[g * 3 + 2];
    _Float16* xt = &xs[w][r * 96 + 64];
    store4h(xt, rx, ry, rz, 0.f);
    store4h(xt + 4, 0, 0, 0, 0);
    store4h(xt + 8, 0, 0, 0, 0);
    store4h(xt + 12, 0, 0, 0, 0);
  } else {
    _Float16* xt = &xs[w][r * 96 + 80];
    store4h(xt, 0, 0, 0, 0);      store4h(xt + 4, 0, 0, 0, 0);
    store4h(xt + 8, 0, 0, 0, 0);  store4h(xt + 12, 0, 0, 0, 0);
  }
  asm volatile("s_wait_dscnt 0x0" ::: "memory");
  v8f y[4];
  mlp2<3>(xs[w], 96, hs[w], wf1, b1, wf2, b2, y);
  int n = l & 15;
#pragma unroll
  for (int i = 0; i < 8; ++i) {
    int dd = lab[w][i + 8 * hf];
    if (dd >= 0) {
#pragma unroll
      for (int t = 0; t < 4; ++t)
        atomicMax((int*)&f3[dd * 64 + t * 16 + n], __float_as_int(y[t][i]));
    }
  }
}

// ---- l2m: MLP([f4[l2l[r]], l1c[r]-l2c[l2l[r]]]) -> f5[r] -----------------
__global__ void k_l2m(const float* __restrict__ f4, const float* __restrict__ l1c,
                      const float* __restrict__ l2c, const int* __restrict__ l2l,
                      int N, const _Float16* wf1, const float* b1,
                      const _Float16* wf2, const float* b2, float* f5) {
  __shared__ _Float16 xs[WPB][16 * 96];
  __shared__ _Float16 hs[WPB][16 * 64];
  int w = threadIdx.x >> 5;
  int tile = blockIdx.x * WPB + w;
  if (tile * 16 >= N) return;
  int l = threadIdx.x & 31, r = l & 15, hf = l >> 4;
  int idx = tile * 16 + r;
  int ic = idx < N ? idx : N - 1;
  int g = l2l[ic];
  const float4* fp = (const float4*)(f4 + g * 64) + hf * 8;
  _Float16* xr = &xs[w][r * 96 + hf * 32];
#pragma unroll
  for (int q = 0; q < 8; ++q) {
    float4 v = fp[q];
    store4h(xr + 4 * q, v.x, v.y, v.z, v.w);
  }
  if (hf == 0) {
    float rx = l1c[ic * 3 + 0] - l2c[g * 3 + 0];
    float ry = l1c[ic * 3 + 1] - l2c[g * 3 + 1];
    float rz = l1c[ic * 3 + 2] - l2c[g * 3 + 2];
    _Float16* xt = &xs[w][r * 96 + 64];
    store4h(xt, rx, ry, rz, 0.f);
    store4h(xt + 4, 0, 0, 0, 0);
    store4h(xt + 8, 0, 0, 0, 0);
    store4h(xt + 12, 0, 0, 0, 0);
  } else {
    _Float16* xt = &xs[w][r * 96 + 80];
    store4h(xt, 0, 0, 0, 0);      store4h(xt + 4, 0, 0, 0, 0);
    store4h(xt + 8, 0, 0, 0, 0);  store4h(xt + 12, 0, 0, 0, 0);
  }
  asm volatile("s_wait_dscnt 0x0" ::: "memory");
  v8f y[4];
  mlp2<3>(xs[w], 96, hs[w], wf1, b1, wf2, b2, y);
  int n = l & 15;
#pragma unroll
  for (int i = 0; i < 8; ++i) {
    int row = tile * 16 + i + 8 * hf;
    if (row < N) {
#pragma unroll
      for (int t = 0; t < 4; ++t)
        f5[row * 64 + t * 16 + n] = y[t][i];
    }
  }
}

// ---- plain GNN on l2: elementwise atomic max of f3[src] into agg ---------
__global__ void k_plainmax(const float* __restrict__ f, const int* __restrict__ edges,
                           int M, float* agg) {
  int id = blockIdx.x * blockDim.x + threadIdx.x;
  if (id >= M * 64) return;
  int e = id >> 6, c = id & 63;
  int s = edges[2 * e], d = edges[2 * e + 1];
  atomicMax((int*)&agg[d * 64 + c], __float_as_int(f[s * 64 + c]));
}

__global__ void k_addres(const float* __restrict__ a, const float* __restrict__ b,
                         int n, float* o) {
  int id = blockIdx.x * blockDim.x + threadIdx.x;
  if (id < n) o[id] = a[id] + b[id];
}

// ---- final: fbn MLP + classifier (64->20) per point -> d_out -------------
__global__ void k_final(const float* __restrict__ f6, const float* __restrict__ pts,
                        const float* __restrict__ l1c, const int* __restrict__ l1l,
                        int N, const _Float16* wf1, const float* b1,
                        const _Float16* wf2, const float* b2,
                        const _Float16* wfc, const float* bc, float* out) {
  __shared__ _Float16 xs[WPB][16 * 96];
  __shared__ _Float16 hs[WPB][16 * 64];
  int w = threadIdx.x >> 5;
  int tile = blockIdx.x * WPB + w;
  if (tile * 16 >= N) return;
  int l = threadIdx.x & 31, r = l & 15, hf = l >> 4;
  int idx = tile * 16 + r;
  int ic = idx < N ? idx : N - 1;
  int li = l1l[ic];
  const float4* fp = (const float4*)(f6 + li * 64) + hf * 8;
  _Float16* xr = &xs[w][r * 96 + hf * 32];
#pragma unroll
  for (int q = 0; q < 8; ++q) {
    float4 v = fp[q];
    store4h(xr + 4 * q, v.x, v.y, v.z, v.w);
  }
  if (hf == 0) {
    float rx = pts[ic * 3 + 0] - l1c[li * 3 + 0];
    float ry = pts[ic * 3 + 1] - l1c[li * 3 + 1];
    float rz = pts[ic * 3 + 2] - l1c[li * 3 + 2];
    _Float16* xt = &xs[w][r * 96 + 64];
    store4h(xt, rx, ry, rz, 0.f);
    store4h(xt + 4, 0, 0, 0, 0);
    store4h(xt + 8, 0, 0, 0, 0);
    store4h(xt + 12, 0, 0, 0, 0);
  } else {
    _Float16* xt = &xs[w][r * 96 + 80];
    store4h(xt, 0, 0, 0, 0);      store4h(xt + 4, 0, 0, 0, 0);
    store4h(xt + 8, 0, 0, 0, 0);  store4h(xt + 12, 0, 0, 0, 0);
  }
  asm volatile("s_wait_dscnt 0x0" ::: "memory");
  v8f y[4];
  mlp2<3>(xs[w], 96, hs[w], wf1, b1, wf2, b2, y);
  int n = l & 15;
  // stage f7 back into LDS as A matrix for the classifier matmul
#pragma unroll
  for (int t = 0; t < 4; ++t)
#pragma unroll
    for (int i = 0; i < 8; ++i)
      hs[w][(i + 8 * hf) * 64 + t * 16 + n] = (_Float16)y[t][i];
  asm volatile("s_wait_dscnt 0x0" ::: "memory");
  v8f z[2];
  z[0] = zero8(); z[1] = zero8();
#pragma unroll
  for (int c = 0; c < 2; ++c) {
    v16h a = load_afrag(hs[w], 64, c);
#pragma unroll
    for (int t = 0; t < 2; ++t) {
      v16h b = *(const v16h*)(wfc + (((c * 2 + t) * 32 + l) << 4));
      z[t] = wmma16(a, b, z[t]);
    }
  }
#pragma unroll
  for (int t = 0; t < 2; ++t) {
#pragma unroll
    for (int i = 0; i < 8; ++i) {
      int col = t * 16 + n;
      int row = tile * 16 + i + 8 * hf;
      if (col < 20 && row < N) out[row * 20 + col] = z[t][i] + bc[col];
    }
  }
}

// ===========================================================================
extern "C" void kernel_launch(void* const* d_in, const int* in_sizes, int n_in,
                              void* d_out, int out_size, void* d_ws, size_t ws_size,
                              hipStream_t stream) {
  (void)out_size; (void)ws_size;
  const int NPTS = 600000, NL1 = 60000, NL2 = 6000, DEG = 16;
  const int M1 = NL1 * DEG, M2 = NL2 * DEG;
  auto F = [&](int i) { return (const float*)d_in[i]; };
  auto I = [&](int i) { return (const int*)d_in[i]; };

  // canonical mlp order: 0 ffn, 1 g2e, 2 g2o, 3 m2l, 4 l2m, 5 g6e, 6 g6o, 7 fbn
  struct PP { const float *w1, *b1, *w2, *b2; };
  PP mp[8];
  const float *clsw = nullptr, *clsb = nullptr;
  const float *rem, *pts, *l1c, *l2c;
  const int *l1e, *l2e, *l1l, *l2l;

  int s0 = (n_in > 0) ? in_sizes[0] : 0;
  if (s0 == 180000) {
    // everything flattened in fully sorted (jax) order incl. top level
    l1c = F(0); l1e = I(1); l1l = I(2); l2c = F(3); l2e = I(4); l2l = I(5);
    pts = F(40); rem = F(41);
    int b = 6;
    clsb = F(b + 0); clsw = F(b + 1);
    auto S = [&](PP& m, int i) { m.b1 = F(b + i); m.w1 = F(b + i + 1);
                                 m.b2 = F(b + i + 2); m.w2 = F(b + i + 3); };
    S(mp[7], 2); S(mp[0], 6); S(mp[1], 10); S(mp[2], 14);
    S(mp[5], 18); S(mp[6], 22); S(mp[4], 26); S(mp[3], 30);
  } else {
    int t0 = n_in - 8;  // tail arrays in setup insertion order
    rem = F(t0 + 0); pts = F(t0 + 1); l1c = F(t0 + 2); l2c = F(t0 + 3);
    l1e = I(t0 + 4); l2e = I(t0 + 5); l1l = I(t0 + 6); l2l = I(t0 + 7);
    if (s0 == 20) {  // params sorted (jax tree): cls,fbn,ffn,g2e,g2o,g6e,g6o,l2m,m2l
      clsb = F(0); clsw = F(1);
      auto S = [&](PP& m, int i) { m.b1 = F(i); m.w1 = F(i + 1);
                                   m.b2 = F(i + 2); m.w2 = F(i + 3); };
      S(mp[7], 2); S(mp[0], 6); S(mp[1], 10); S(mp[2], 14);
      S(mp[5], 18); S(mp[6], 22); S(mp[4], 26); S(mp[3], 30);
    } else if (s0 == 64) {  // insertion top-level, sorted within dense (b,w)
      for (int k = 0; k < 8; ++k) {
        mp[k].b1 = F(4 * k); mp[k].w1 = F(4 * k + 1);
        mp[k].b2 = F(4 * k + 2); mp[k].w2 = F(4 * k + 3);
      }
      clsb = F(32); clsw = F(33);
    } else {  // s0 == 256: pure insertion order (w before b)
      for (int k = 0; k < 8; ++k) {
        mp[k].w1 = F(4 * k); mp[k].b1 = F(4 * k + 1);
        mp[k].w2 = F(4 * k + 2); mp[k].b2 = F(4 * k + 3);
      }
      clsw = F(32); clsb = F(33);
    }
  }

  // workspace layout
  char* ws = (char*)d_ws;
  size_t off = 0;
  auto alloc = [&](size_t bytes) {
    size_t r = off; off = (off + bytes + 255) & ~(size_t)255; return r;
  };
  size_t fsz = (size_t)NL1 * 64 * 4;
  float* f1  = (float*)(ws + alloc(fsz));
  float* f2  = (float*)(ws + alloc(fsz));
  float* agg = (float*)(ws + alloc(fsz));
  float* f6  = (float*)(ws + alloc(fsz));
  float* f3  = (float*)(ws + alloc((size_t)NL2 * 64 * 4));
  float* f4  = (float*)(ws + alloc((size_t)NL2 * 64 * 4));
  float* f5  = f1;  // f1 dead after gnn1 output

  int ch1[8]  = {1, 3, 2, 3, 3, 3, 2, 3};   // K-chunks of layer 1
  int din1[8] = {4, 67, 64, 67, 67, 67, 64, 67};
  _Float16* wf1[8]; _Float16* wf2[8];
  for (int k = 0; k < 8; ++k) {
    wf1[k] = (_Float16*)(ws + alloc((size_t)ch1[k] * 4 * 512 * 2));
    wf2[k] = (_Float16*)(ws + alloc((size_t)2 * 4 * 512 * 2));
  }
  _Float16* wfc = (_Float16*)(ws + alloc((size_t)2 * 2 * 512 * 2));

  // pack all weights into WMMA B-fragment layout
  for (int k = 0; k < 8; ++k) {
    int tot1 = ch1[k] * 4 * 512;
    k_pack<<<(tot1 + 255) / 256, 256, 0, stream>>>(mp[k].w1, wf1[k], din1[k], 64,
                                                   ch1[k], 4);
    k_pack<<<(2 * 4 * 512 + 255) / 256, 256, 0, stream>>>(mp[k].w2, wf2[k], 64, 64,
                                                          2, 4);
  }
  k_pack<<<(2 * 2 * 512 + 255) / 256, 256, 0, stream>>>(clsw, wfc, 64, 20, 2, 2);

  // stage 1: f1 = segment_sum(mlp_ffn(x), l1_labels)
  hipMemsetAsync(f1, 0, fsz, stream);
  {
    int tiles = (NPTS + 15) / 16;
    k_ffn<<<(tiles + WPB - 1) / WPB, 32 * WPB, 0, stream>>>(
        pts, rem, l1c, l1l, NPTS, wf1[0], mp[0].b1, wf2[0], mp[0].b2, f1);
  }

  // gnn1: f2 = f1 + mlp_g2o(seg_max(mlp_g2e(edge feats)))
  hipMemsetAsync(agg, 0, fsz, stream);
  {
    int tiles = (M1 + 15) / 16;
    k_edge<<<(tiles + WPB - 1) / WPB, 32 * WPB, 0, stream>>>(
        f1, l1c, l1e, M1, wf1[1], mp[1].b1, wf2[1], mp[1].b2, agg);
  }
  {
    int tiles = (NL1 + 15) / 16;
    k_gnnout<<<(tiles + WPB - 1) / WPB, 32 * WPB, 0, stream>>>(
        f1, agg, NL1, wf1[2], mp[2].b1, wf2[2], mp[2].b2, nullptr, f2);
  }

  // f3 = seg_max(mlp_m2l([f2, rel]), l2_labels)
  hipMemsetAsync(f3, 0, (size_t)NL2 * 64 * 4, stream);
  {
    int tiles = (NL1 + 15) / 16;
    k_m2l<<<(tiles + WPB - 1) / WPB, 32 * WPB, 0, stream>>>(
        f2, l1c, l2c, l2l, NL1, wf1[3], mp[3].b1, wf2[3], mp[3].b2, f3);
  }

  // f4 = f3 + seg_max(f3[src], dst)  (plain gnn, reuse agg region)
  hipMemsetAsync(agg, 0, (size_t)NL2 * 64 * 4, stream);
  k_plainmax<<<(M2 * 64 + 255) / 256, 256, 0, stream>>>(f3, l2e, M2, agg);
  k_addres<<<(NL2 * 64 + 255) / 256, 256, 0, stream>>>(f3, agg, NL2 * 64, f4);

  // f5 = mlp_l2m([f4[l2_labels], rel])
  {
    int tiles = (NL1 + 15) / 16;
    k_l2m<<<(tiles + WPB - 1) / WPB, 32 * WPB, 0, stream>>>(
        f4, l1c, l2c, l2l, NL1, wf1[4], mp[4].b1, wf2[4], mp[4].b2, f5);
  }

  // gnn2: f6 = f5 + mlp_g6o(seg_max(mlp_g6e(...))) + f2
  hipMemsetAsync(agg, 0, fsz, stream);
  {
    int tiles = (M1 + 15) / 16;
    k_edge<<<(tiles + WPB - 1) / WPB, 32 * WPB, 0, stream>>>(
        f5, l1c, l1e, M1, wf1[5], mp[5].b1, wf2[5], mp[5].b2, agg);
  }
  {
    int tiles = (NL1 + 15) / 16;
    k_gnnout<<<(tiles + WPB - 1) / WPB, 32 * WPB, 0, stream>>>(
        f5, agg, NL1, wf1[6], mp[6].b1, wf2[6], mp[6].b2, f2, f6);
  }

  // final: out = mlp_fbn([f6[l1_labels], rel]) @ cls.w + cls.b
  {
    int tiles = (NPTS + 15) / 16;
    k_final<<<(tiles + WPB - 1) / WPB, 32 * WPB, 0, stream>>>(
        f6, pts, l1c, l1l, NPTS, wf1[7], mp[7].b1, wf2[7], mp[7].b2, wfc, clsb,
        (float*)d_out);
  }
}
